// Beit_6743098655015
// MI455X (gfx1250) — matrix operand order repaired and checked by reading, not compiled, not run on hardware
//
#include <hip/hip_runtime.h>

// ---------------------------------------------------------------------------
// BEiT-style video window attention for MI455X (gfx1250, wave32, WMMA).
// All 4 matmuls on v_wmma_f32_16x16x32_f16 (f16 in, f32 accumulate).
// Attention k/v tiles staged to LDS by the Tensor Data Mover (double-buffered,
// s_wait_tensorcnt); GEMM K-loops manually 2x-unrolled ping-pong pipelined.
// ---------------------------------------------------------------------------

typedef _Float16 h16;
typedef __attribute__((ext_vector_type(16))) _Float16 v16h;
typedef __attribute__((ext_vector_type(8)))  float    v8f;
typedef unsigned int u32;
typedef unsigned long long u64;
typedef u32 __attribute__((ext_vector_type(4))) u32x4;
typedef int __attribute__((ext_vector_type(8))) i32x8;
typedef int __attribute__((ext_vector_type(4))) i32x4;

#define NUM_HEADS 12
#define NFRAMES   8
#define SEQ       1568          // N
#define CDIM      768           // C
#define HDIM      64            // head dim
#define BATCH     2
#define TPF       196           // tokens per frame
#define POSD      197           // spatial bias period
#define NPOS      (POSD*POSD)   // 38809
#define BN        (BATCH*SEQ)   // 3136

#if defined(__HIP_DEVICE_COMPILE__) && __has_builtin(__builtin_amdgcn_s_wait_tensorcnt)
#define WAIT_TDM(n) __builtin_amdgcn_s_wait_tensorcnt((short)(n))
#else
#define WAIT_TDM(n) do {} while (0)
#endif

static __device__ __forceinline__ v8f zero8() {
  v8f z;
#pragma unroll
  for (int i = 0; i < 8; ++i) z[i] = 0.0f;
  return z;
}

static __device__ __forceinline__ v8f wmma16(v16h a, v16h b, v8f c) {
  return __builtin_amdgcn_wmma_f32_16x16x32_f16(
      false, a, false, b, (short)0, c, false, false);
}

// A fragment 16x32 (MxK), row-major source, K contiguous.
// lane<16: row M=lane, K=0..15; lane>=16: row M=lane-16, K=16..31.
static __device__ __forceinline__ v16h load_frag_a(const h16* src, int row_stride, int lane) {
  const int m    = lane & 15;
  const int koff = (lane >> 4) * 16;
  const h16* p = src + (size_t)m * row_stride + koff;
  v16h a;
#pragma unroll
  for (int i = 0; i < 8; ++i) a[i] = p[i];
#pragma unroll
  for (int i = 0; i < 8; ++i) a[8 + i] = p[8 + i];
  return a;
}

// B fragment 32x16 (KxN): B[k][n] = src[n*row_stride + k].
// lane<16: col N=lane, K=0..15; lane>=16: col N=lane-16, K=16..31.
static __device__ __forceinline__ v16h load_frag_b(const h16* src, int row_stride, int lane) {
  const int n    = lane & 15;
  const int koff = (lane >> 4) * 16;
  const h16* p = src + (size_t)n * row_stride + koff;
  v16h b;
#pragma unroll
  for (int i = 0; i < 16; ++i) b[i] = p[i];
  return b;
}

// ---------------------------------------------------------------------------
// Tensor Data Mover: async copy of a 2D f16 tile (tile_d1 rows x tile_d0 cols,
// row pitch d0_stride elements) from global into LDS (packed row-major).
// D# packing per CDNA5 ISA. Tensor dims == tile dims
// (tile known in-bounds), groups 2/3/4 zero. 6-arg builtin on this toolchain.
// ---------------------------------------------------------------------------
static __device__ __forceinline__ void tdm_load_tile_2d(
    h16* lds_dst, const h16* gsrc, u32 tile_d0, u32 tile_d1,
    u32 d0_stride, int lane) {
#if defined(__HIP_DEVICE_COMPILE__) && __has_builtin(__builtin_amdgcn_tensor_load_to_lds)
  (void)lane;
  const u32 lds_off = (u32)(size_t)lds_dst;       // generic addr low 32 = LDS offset
  const u64 ga = (u64)(size_t)gsrc;
  u32x4 g0;
  g0[0] = 1u;                                     // count=1 (valid user descriptor)
  g0[1] = lds_off;                                // lds_addr
  g0[2] = (u32)ga;                                // global_addr[31:0]
  g0[3] = (u32)(ga >> 32) | (2u << 30);           // global_addr[56:32] | type=2
  i32x8 g1;
  g1[0] = (int)(1u << 16);                        // data_size=1 -> 2 bytes; mask/flags 0
  g1[1] = (int)((tile_d0 & 0xFFFFu) << 16);       // tensor_dim0[15:0]
  g1[2] = (int)((tile_d0 >> 16) | ((tile_d1 & 0xFFFFu) << 16)); // d0 hi | tensor_dim1 lo
  g1[3] = (int)((tile_d1 >> 16) | (tile_d0 << 16));             // d1 hi | tile_dim0
  g1[4] = (int)(tile_d1 & 0xFFFFu);               // tile_dim1 | tile_dim2=0
  g1[5] = (int)d0_stride;                         // tensor_dim0_stride[31:0]
  g1[6] = 0;                                      // stride0 hi | stride1 lo
  g1[7] = 0;                                      // stride1 hi
  i32x4 g2; g2[0] = 0; g2[1] = 0; g2[2] = 0; g2[3] = 0;
  i32x4 g3; g3[0] = 0; g3[1] = 0; g3[2] = 0; g3[3] = 0;
  i32x8 g4;
#pragma unroll
  for (int i = 0; i < 8; ++i) g4[i] = 0;
  __builtin_amdgcn_tensor_load_to_lds(g0, g1, g2, g3, g4, 0);
#else
  // synchronous wave-cooperative fallback
  const u32 total = tile_d0 * tile_d1;
  for (u32 e = (u32)lane; e < total; e += 32) {
    const u32 r = e / tile_d0, cidx = e % tile_d0;
    lds_dst[e] = gsrc[(size_t)r * d0_stride + cidx];
  }
#endif
}

// ---------------------------------------------------------------------------
// Prep kernels
// ---------------------------------------------------------------------------
__global__ void cvt_f32_f16(const float* __restrict__ s, h16* __restrict__ d, int n) {
  int i = blockIdx.x * blockDim.x + threadIdx.x;
  if (i < n) d[i] = (h16)s[i];
}

__global__ void gather_pos(const float* __restrict__ table, const int* __restrict__ idx,
                           float* __restrict__ posg) {
  int e = blockIdx.x * blockDim.x + threadIdx.x;
  if (e < NPOS) {
    int id = idx[e];
#pragma unroll
    for (int h = 0; h < NUM_HEADS; ++h)
      posg[(size_t)h * NPOS + e] = table[id * NUM_HEADS + h];
  }
}

__global__ void gather_temp(const float* __restrict__ table, const int* __restrict__ idx,
                            float* __restrict__ tempg) {
  int e = threadIdx.x;
  if (e < NFRAMES * NFRAMES) {
    int id = idx[e];
#pragma unroll
    for (int h = 0; h < NUM_HEADS; ++h)
      tempg[h * NFRAMES * NFRAMES + e] = table[id * NUM_HEADS + h];
  }
}

// ---------------------------------------------------------------------------
// Shared GEMM core: 16x64 f32 tile = A[16xK] * B[64xK]^T, K=CDIM, ping-pong
// 2x-unrolled pipeline (no register rotation moves).
// ---------------------------------------------------------------------------
static __device__ __forceinline__ void gemm_16x64(
    const h16* __restrict__ ap, const h16* __restrict__ bp, int lane, v8f c[4]) {
#pragma unroll
  for (int f = 0; f < 4; ++f) c[f] = zero8();

  v16h a0 = load_frag_a(ap, CDIM, lane);
  v16h b0[4];
#pragma unroll
  for (int f = 0; f < 4; ++f) b0[f] = load_frag_b(bp + (size_t)f * 16 * CDIM, CDIM, lane);

  for (int kk = 0; kk < CDIM; kk += 64) {
    // stage-1 prefetch (iteration-local)
    v16h a1 = load_frag_a(ap + kk + 32, CDIM, lane);
    v16h b1[4];
#pragma unroll
    for (int f = 0; f < 4; ++f)
      b1[f] = load_frag_b(bp + (size_t)f * 16 * CDIM + kk + 32, CDIM, lane);
#pragma unroll
    for (int f = 0; f < 4; ++f) c[f] = wmma16(a0, b0[f], c[f]);
    // stage-0 prefetch for next iteration (same names -> no rotation)
    if (kk + 64 < CDIM) {
      a0 = load_frag_a(ap + kk + 64, CDIM, lane);
#pragma unroll
      for (int f = 0; f < 4; ++f)
        b0[f] = load_frag_b(bp + (size_t)f * 16 * CDIM + kk + 64, CDIM, lane);
    }
#pragma unroll
    for (int f = 0; f < 4; ++f) c[f] = wmma16(a1, b1[f], c[f]);
  }
}

// ---------------------------------------------------------------------------
// QKV GEMM: out[i][j] = sum_c x[i][c] * qkv_w[j][c] + bias(j); 16x64 tile/wave.
// Epilogue scatters q (scaled) / k / v^T as f16.
// ---------------------------------------------------------------------------
__global__ __launch_bounds__(32) void qkv_gemm(
    const h16* __restrict__ xh, const h16* __restrict__ wh,
    const float* __restrict__ q_bias, const float* __restrict__ v_bias,
    h16* __restrict__ q_s, h16* __restrict__ k_s, h16* __restrict__ vT) {
  const int lane  = threadIdx.x;
  const int ibase = blockIdx.x * 16;
  const int jbase = blockIdx.y * 64;

  v8f c[4];
  gemm_16x64(xh + (size_t)ibase * CDIM, wh + (size_t)jbase * CDIM, lane, c);

  const int half8 = (lane & 16) ? 8 : 0;
  const int col   = lane & 15;
#pragma unroll
  for (int f = 0; f < 4; ++f) {
    const int j     = jbase + f * 16 + col;
    const int which = j / CDIM;       // 0=q, 1=k, 2=v
    const int jr    = j % CDIM;
    const int h     = jr / HDIM;
    const int d     = jr % HDIM;
    const float bias = (which == 0) ? q_bias[jr] : ((which == 2) ? v_bias[jr] : 0.0f);
#pragma unroll
    for (int r = 0; r < 8; ++r) {
      const int i = ibase + r + half8;
      const int b = i / SEQ;
      const int n = i % SEQ;
      const float val = c[f][r] + bias;
      const size_t bh = (size_t)b * NUM_HEADS + h;
      if (which == 0)      q_s[(bh * SEQ + n) * HDIM + d] = (h16)(val * 0.125f);
      else if (which == 1) k_s[(bh * SEQ + n) * HDIM + d] = (h16)val;
      else                 vT[(bh * HDIM + d) * SEQ + n]  = (h16)val;
    }
  }
}

// ---------------------------------------------------------------------------
// Fused flash attention. One wave per 16-row q tile per (b,h). k/v tiles are
// TDM-staged into double-buffered LDS; 4 WMMAs QK^T + bias + online softmax +
// 4 WMMAs PV per 32-key chunk.
// ---------------------------------------------------------------------------
__global__ __launch_bounds__(32) void attn_fused(
    const h16* __restrict__ q_s, const h16* __restrict__ k_s, const h16* __restrict__ vT,
    const float* __restrict__ posg, const float* __restrict__ tempg,
    h16* __restrict__ attnh) {
  const int lane = threadIdx.x;
  const int qt = blockIdx.x;            // 0..97
  const int h  = blockIdx.y;
  const int b  = blockIdx.z;
  const size_t bh = (size_t)b * NUM_HEADS + h;

  __shared__ h16 kbuf[2][32 * HDIM];    // [keys][feat]
  __shared__ h16 vbuf[2][HDIM * 32];    // [feat][keys]
  __shared__ h16 pbuf[16 * 32];

  const h16* kg = k_s + bh * SEQ * HDIM;   // [1568][64]
  const h16* vg = vT  + bh * HDIM * SEQ;   // [64][1568]

  const h16* qb = q_s + (bh * SEQ + (size_t)qt * 16) * HDIM;
  const v16h aq0 = load_frag_a(qb,      HDIM, lane);
  const v16h aq1 = load_frag_a(qb + 32, HDIM, lane);

  const int half8 = (lane & 16) ? 8 : 0;
  const int col   = lane & 15;

  int qm[8], qf[8];
#pragma unroll
  for (int r = 0; r < 8; ++r) {
    const int qq = qt * 16 + r + half8;
    qm[r] = qq % POSD;
    qf[r] = qq / TPF;
  }
  const float* posg_h = posg + (size_t)h * NPOS;
  const float* temp_h = tempg + h * NFRAMES * NFRAMES;

  float m[8], l[8];
#pragma unroll
  for (int r = 0; r < 8; ++r) { m[r] = -1e30f; l[r] = 0.0f; }
  v8f acc[4];
#pragma unroll
  for (int f = 0; f < 4; ++f) acc[f] = zero8();

  const int nchunks = SEQ / 32;         // 49
  // prologue: stage chunk 0
  tdm_load_tile_2d(kbuf[0], kg, HDIM, 32, HDIM, lane);
  tdm_load_tile_2d(vbuf[0], vg, 32, HDIM, SEQ, lane);

#pragma unroll 2
  for (int ci = 0; ci < nchunks; ++ci) {
    const int kb  = ci * 32;
    const int cur = ci & 1;
    const int nxt = cur ^ 1;
    if (ci + 1 < nchunks) {
      tdm_load_tile_2d(kbuf[nxt], kg + (size_t)(kb + 32) * HDIM, HDIM, 32, HDIM, lane);
      tdm_load_tile_2d(vbuf[nxt], vg + kb + 32, 32, HDIM, SEQ, lane);
      WAIT_TDM(2);                      // current buffer's 2 transfers done
    } else {
      WAIT_TDM(0);
    }

    v8f s[2];
#pragma unroll
    for (int t = 0; t < 2; ++t) {
      const h16* kt = kbuf[cur] + (size_t)t * 16 * HDIM;
      const v16h bk0 = load_frag_b(kt,      HDIM, lane);
      const v16h bk1 = load_frag_b(kt + 32, HDIM, lane);
      s[t] = wmma16(aq0, bk0, zero8());
      s[t] = wmma16(aq1, bk1, s[t]);
      const int key = kb + t * 16 + col;
      const int km  = key % POSD;
      const int kf  = key / TPF;
#pragma unroll
      for (int r = 0; r < 8; ++r)
        s[t][r] += posg_h[qm[r] * POSD + km] + temp_h[qf[r] * NFRAMES + kf];
    }

    // online softmax: rows across VGPRs, keys across 16-lane halves
#pragma unroll
    for (int r = 0; r < 8; ++r) {
      float rmax = fmaxf(s[0][r], s[1][r]);
#pragma unroll
      for (int off = 1; off <= 8; off <<= 1)
        rmax = fmaxf(rmax, __shfl_xor(rmax, off, 32));
      const float mn    = fmaxf(m[r], rmax);
      const float alpha = __expf(m[r] - mn);
      const float p0 = __expf(s[0][r] - mn);
      const float p1 = __expf(s[1][r] - mn);
      float rsum = p0 + p1;
#pragma unroll
      for (int off = 1; off <= 8; off <<= 1)
        rsum += __shfl_xor(rsum, off, 32);
      l[r] = l[r] * alpha + rsum;
      m[r] = mn;
      s[0][r] = p0; s[1][r] = p1;
#pragma unroll
      for (int f = 0; f < 4; ++f) acc[f][r] *= alpha;
    }

    // P -> LDS bounce, re-enter as A fragment for PV
    __syncthreads();
#pragma unroll
    for (int t = 0; t < 2; ++t)
#pragma unroll
      for (int r = 0; r < 8; ++r)
        pbuf[(r + half8) * 32 + t * 16 + col] = (h16)s[t][r];
    __syncthreads();

    const v16h apf = load_frag_a(pbuf, 32, lane);
#pragma unroll
    for (int f = 0; f < 4; ++f) {
      const v16h bv = load_frag_b(vbuf[cur] + (size_t)f * 16 * 32, 32, lane);
      acc[f] = wmma16(apf, bv, acc[f]);
    }
    __syncthreads();
  }

  // normalize (nan_to_num guard) and store f16 [b, n, h*64+d]
#pragma unroll
  for (int r = 0; r < 8; ++r) {
    const float invl = (l[r] > 0.0f) ? 1.0f / l[r] : 0.0f;
    const int row = qt * 16 + r + half8;
    const size_t obase = ((size_t)b * SEQ + row) * CDIM + h * HDIM;
#pragma unroll
    for (int f = 0; f < 4; ++f)
      attnh[obase + f * 16 + col] = (h16)(acc[f][r] * invl);
  }
}

// ---------------------------------------------------------------------------
// Output projection: out[i][j] = sum_c attn[i][c] * proj_w[j][c] + proj_b[j]
// ---------------------------------------------------------------------------
__global__ __launch_bounds__(32) void proj_gemm(
    const h16* __restrict__ ah, const h16* __restrict__ wh,
    const float* __restrict__ bias, float* __restrict__ out) {
  const int lane  = threadIdx.x;
  const int ibase = blockIdx.x * 16;
  const int jbase = blockIdx.y * 64;

  v8f c[4];
  gemm_16x64(ah + (size_t)ibase * CDIM, wh + (size_t)jbase * CDIM, lane, c);

  const int half8 = (lane & 16) ? 8 : 0;
  const int col   = lane & 15;
#pragma unroll
  for (int f = 0; f < 4; ++f) {
    const int j = jbase + f * 16 + col;
    const float bj = bias[j];
#pragma unroll
    for (int r = 0; r < 8; ++r) {
      const int i = ibase + r + half8;
      out[(size_t)i * CDIM + j] = c[f][r] + bj;
    }
  }
}

// ---------------------------------------------------------------------------
// Host launcher (graph-capture safe; all scratch from d_ws, ~31 MB)
// ---------------------------------------------------------------------------
extern "C" void kernel_launch(void* const* d_in, const int* in_sizes, int n_in,
                              void* d_out, int out_size, void* d_ws, size_t ws_size,
                              hipStream_t stream) {
  (void)in_sizes; (void)n_in; (void)out_size; (void)ws_size;
  const float* x              = (const float*)d_in[0];
  const float* qkv_w          = (const float*)d_in[1];
  const float* q_bias         = (const float*)d_in[2];
  const float* v_bias         = (const float*)d_in[3];
  const float* rel_pos_table  = (const float*)d_in[4];
  const float* rel_temp_table = (const float*)d_in[5];
  const float* proj_w         = (const float*)d_in[6];
  const float* proj_b         = (const float*)d_in[7];
  const int*   rel_pos_index  = (const int*)d_in[8];
  const int*   rel_temp_index = (const int*)d_in[9];
  float* out = (float*)d_out;

  char* ws = (char*)d_ws;
  auto carve = [&](size_t bytes) {
    char* p = ws;
    ws += (bytes + 255) & ~(size_t)255;
    return p;
  };
  h16*   xh     = (h16*)carve((size_t)BN * CDIM * sizeof(h16));
  h16*   qkvwh  = (h16*)carve((size_t)3 * CDIM * CDIM * sizeof(h16));
  h16*   projwh = (h16*)carve((size_t)CDIM * CDIM * sizeof(h16));
  h16*   q_s    = (h16*)carve((size_t)BATCH * NUM_HEADS * SEQ * HDIM * sizeof(h16));
  h16*   k_s    = (h16*)carve((size_t)BATCH * NUM_HEADS * SEQ * HDIM * sizeof(h16));
  h16*   vT     = (h16*)carve((size_t)BATCH * NUM_HEADS * SEQ * HDIM * sizeof(h16));
  h16*   attnh  = (h16*)carve((size_t)BN * CDIM * sizeof(h16));
  float* posg   = (float*)carve((size_t)NUM_HEADS * NPOS * sizeof(float));
  float* tempg  = (float*)carve((size_t)NUM_HEADS * NFRAMES * NFRAMES * sizeof(float));

  int n;
  n = BN * CDIM;
  cvt_f32_f16<<<(n + 255) / 256, 256, 0, stream>>>(x, xh, n);
  n = 3 * CDIM * CDIM;
  cvt_f32_f16<<<(n + 255) / 256, 256, 0, stream>>>(qkv_w, qkvwh, n);
  n = CDIM * CDIM;
  cvt_f32_f16<<<(n + 255) / 256, 256, 0, stream>>>(proj_w, projwh, n);
  gather_pos<<<(NPOS + 255) / 256, 256, 0, stream>>>(rel_pos_table, rel_pos_index, posg);
  gather_temp<<<1, 64, 0, stream>>>(rel_temp_table, rel_temp_index, tempg);

  qkv_gemm<<<dim3(BN / 16, (3 * CDIM) / 64), 32, 0, stream>>>(
      xh, qkvwh, q_bias, v_bias, q_s, k_s, vT);

  attn_fused<<<dim3(SEQ / 16, NUM_HEADS, BATCH), 32, 0, stream>>>(
      q_s, k_s, vT, posg, tempg, attnh);

  proj_gemm<<<dim3(BN / 16, CDIM / 64), 32, 0, stream>>>(attnh, projwh, proj_b, out);
}
